// SelfAttention_3470333575729
// MI455X (gfx1250) — matrix operand order repaired and checked by reading, not compiled, not run on hardware
//
#include <hip/hip_runtime.h>
#include <hip/hip_bf16.h>
#include <math.h>

typedef __attribute__((ext_vector_type(16))) __bf16 v16bf;
typedef __attribute__((ext_vector_type(8)))  __bf16 v8bf;
typedef __attribute__((ext_vector_type(8)))  float  v8f;
typedef __attribute__((ext_vector_type(4)))  unsigned int v4u;
typedef __attribute__((ext_vector_type(8)))  int v8i;
typedef __attribute__((ext_vector_type(4)))  int v4i;

constexpr int   kB   = 8;
constexpr int   kS   = 2048;
constexpr int   kD   = 256;
constexpr float kEps = 1e-5f;

// ---------------------------------------------------------------------------
// WMMA wrapper: D = A(16x32 bf16) * B(32x16 bf16) + C(16x16 f32)
// ---------------------------------------------------------------------------
__device__ __forceinline__ v8f wmma_bf16(v16bf a, v16bf b, v8f c) {
  return __builtin_amdgcn_wmma_f32_16x16x32_bf16(
      /*neg_a=*/false, a, /*neg_b=*/false, b,
      /*c_mod=*/(short)0, c, /*reuse_a=*/false, /*reuse_b=*/false);
}

// ---------------------------------------------------------------------------
// TDM: async 2D tile DMA global->LDS (CDNA5 ISA ch.8).  Descriptor per
// D# group0/group1 bitfields: count=1, type=2 ("image"), data_size=2 bytes,
// tensor_dim0=width, tensor_dim1=rows, tile = whole region, row stride given.
// Tracked by TENSORcnt (per-wave); EXEC ignored (one DMA per wave).
// ---------------------------------------------------------------------------
__device__ __forceinline__ void tdm_load_2d(const void* gptr, unsigned lds_off,
                                            unsigned width, unsigned rows,
                                            unsigned stride_elems) {
  const unsigned long long ga = (unsigned long long)(uintptr_t)gptr;
  v4u g0;
  g0[0] = 1u;                                           // count=1, user desc
  g0[1] = lds_off;                                      // lds_addr (bytes)
  g0[2] = (unsigned)(ga & 0xFFFFFFFFu);                 // global_addr[31:0]
  g0[3] = (unsigned)((ga >> 32) & 0x01FFFFFFu) | (2u << 30);  // addr[56:32]|type=2
  v8i g1;
  g1[0] = (int)(1u << 16);                              // wg_mask=0, data_size=1 (2B)
  g1[1] = (int)((width & 0xFFFFu) << 16);               // tensor_dim0[15:0]
  g1[2] = (int)((width >> 16) | ((rows & 0xFFFFu) << 16));   // dim0[31:16]|dim1[15:0]
  g1[3] = (int)((rows >> 16) | ((width & 0xFFFFu) << 16));   // dim1[31:16]|tile_dim0
  g1[4] = (int)(rows & 0xFFFFu);                        // tile_dim1, tile_dim2=0
  g1[5] = (int)stride_elems;                            // tensor_dim0_stride[31:0]
  g1[6] = 0;                                            // stride hi, dim1_stride lo
  g1[7] = 0;
  const v4i zz = {0, 0, 0, 0};
#if defined(__clang_major__) && __clang_major__ >= 23
  const v8i zz8 = {0, 0, 0, 0, 0, 0, 0, 0};
  __builtin_amdgcn_tensor_load_to_lds(g0, g1, zz, zz, zz8, 0);
#else
  __builtin_amdgcn_tensor_load_to_lds(g0, g1, zz, zz, 0);
#endif
}

__device__ __forceinline__ unsigned lds_off_of(const void* p) {
  return (unsigned)(uintptr_t)p;   // generic LDS address: offset in low 32 bits
}

// ---------------------------------------------------------------------------
// Fragment loaders (layouts per CDNA5 ISA 7.12.2, wave32).
// A (16x32, MxK): lane l -> row (l&15); K = {kh..kh+7, kh+16..kh+23}, kh=(l>>4)*8
// B (32x16, KxN): lane l -> col (l&15); K = {kh..kh+7, kh+16..kh+23}, kh=(l>>4)*8
// Work from generic pointers: emit global_load_b128 or ds_load_b128 as needed.
// ---------------------------------------------------------------------------
__device__ __forceinline__ v16bf load_frag_a_bf(const __bf16* p, int ld, int lane) {
  const int r  = lane & 15;
  const int kh = (lane >> 4) * 8;
  const __bf16* q = p + (size_t)r * ld + kh;
  v8bf lo = *(const v8bf*)(q);
  v8bf hi = *(const v8bf*)(q + 16);
  v16bf f;
#pragma unroll
  for (int i = 0; i < 8; ++i) { f[i] = lo[i]; f[8 + i] = hi[i]; }
  return f;
}

__device__ __forceinline__ v16bf load_frag_b_bf(const __bf16* p, int ld, int lane) {
  const int r  = lane & 15;
  const int kh = (lane >> 4) * 8;
  const __bf16* q = p + (size_t)r * ld + kh;
  v8bf lo = *(const v8bf*)(q);
  v8bf hi = *(const v8bf*)(q + 16);
  v16bf f;
#pragma unroll
  for (int i = 0; i < 8; ++i) { f[i] = lo[i]; f[8 + i] = hi[i]; }
  return f;
}

// 16-lane reductions (C/D layout: a row's 16 columns live in 16 lanes)
__device__ __forceinline__ float redmax16(float v) {
#pragma unroll
  for (int m = 1; m < 16; m <<= 1) v = fmaxf(v, __shfl_xor(v, m, 32));
  return v;
}
__device__ __forceinline__ float redsum16(float v) {
#pragma unroll
  for (int m = 1; m < 16; m <<= 1) v += __shfl_xor(v, m, 32);
  return v;
}

// ---------------------------------------------------------------------------
// Kernel 0: one-time fp32 -> bf16 conversion (vectorized 8-wide).
// ---------------------------------------------------------------------------
__global__ __launch_bounds__(256) void cvt_bf16_kernel(
    const float* __restrict__ src, __bf16* __restrict__ dst, int n8) {
  const int i = blockIdx.x * blockDim.x + threadIdx.x;
  if (i >= n8) return;
  v8f v = *(const v8f*)(src + (size_t)i * 8);
  v8bf o;
#pragma unroll
  for (int j = 0; j < 8; ++j) o[j] = (__bf16)v[j];
  *(v8bf*)(dst + (size_t)i * 8) = o;
}

// ---------------------------------------------------------------------------
// Kernel 1: fused QKV projection (all-bf16 operands).
// q,k -> bf16 [B,S,D]; v -> bf16 [B,D,S] (transposed for the P*V stage).
// ---------------------------------------------------------------------------
__global__ __launch_bounds__(128) void qkv_kernel(
    const __bf16* __restrict__ xb,
    const __bf16* __restrict__ Wq, const float* __restrict__ bq,
    const __bf16* __restrict__ Wk, const float* __restrict__ bk,
    const __bf16* __restrict__ Wv, const float* __restrict__ bv,
    __bf16* __restrict__ qb, __bf16* __restrict__ kb, __bf16* __restrict__ vT) {
  const int blk  = blockIdx.x;
  const int b    = blk / (kS / 16);
  const int st   = blk % (kS / 16);
  const int wave = threadIdx.x >> 5;
  const int lane = threadIdx.x & 31;
  const int half = lane >> 4;
  const int col  = lane & 15;

  const __bf16* xbase = xb + ((size_t)b * kS + (size_t)st * 16) * kD;

  v16bf xf[kD / 32];
#pragma unroll
  for (int ks = 0; ks < kD / 32; ++ks)
    xf[ks] = load_frag_a_bf(xbase + ks * 32, kD, lane);

  const v8f vzero = {0.f, 0.f, 0.f, 0.f, 0.f, 0.f, 0.f, 0.f};

  for (int et = wave; et < kD / 16; et += 4) {
    const int e0 = et * 16;
    v8f aq = vzero, ak = vzero, av = vzero;
#pragma unroll
    for (int ks = 0; ks < kD / 32; ++ks) {
      v16bf wq = load_frag_b_bf(Wq + (size_t)e0 * kD + ks * 32, kD, lane);
      v16bf wk = load_frag_b_bf(Wk + (size_t)e0 * kD + ks * 32, kD, lane);
      v16bf wv = load_frag_b_bf(Wv + (size_t)e0 * kD + ks * 32, kD, lane);
      aq = wmma_bf16(xf[ks], wq, aq);
      ak = wmma_bf16(xf[ks], wk, ak);
      av = wmma_bf16(xf[ks], wv, av);
    }
    const float biasq = bq[e0 + col];
    const float biask = bk[e0 + col];
    const float biasv = bv[e0 + col];
#pragma unroll
    for (int v = 0; v < 8; ++v) {
      const int s = st * 16 + half * 8 + v;           // absolute sequence row
      const size_t qi = ((size_t)b * kS + s) * kD + e0 + col;
      qb[qi] = (__bf16)(aq[v] + biasq);
      kb[qi] = (__bf16)(ak[v] + biask);
      vT[(size_t)b * kD * kS + (size_t)(e0 + col) * kS + s] = (__bf16)(av[v] + biasv);
    }
  }
}

// ---------------------------------------------------------------------------
// Kernel 2: flash attention with TDM double buffering.
// Block = 64 threads (2 waves) per 16-row q-tile; wave owns 128 ctx features.
//   K tile  (32x256 bf16, shared): wave 0 issues tensor_load_to_lds; its
//     s_wait_tensorcnt + the workgroup barrier publish completion to wave 1.
//   V tile  (128x32 bf16, per wave): each wave issues its own DMA (TENSORcnt
//     is per-wave, so no cross-wave sync needed).
// Next tile's DMAs are issued before computing the current one -> overlap.
// ---------------------------------------------------------------------------
__global__ __launch_bounds__(64) void attn_kernel(
    const __bf16* __restrict__ qb, const __bf16* __restrict__ kb,
    const __bf16* __restrict__ vT, __bf16* __restrict__ ctxb) {
  __shared__ __bf16 kbuf[2][32 * kD];        // 32 KiB: double-buffered K tiles
  __shared__ __bf16 vbuf[2][2][128 * 32];    // 32 KiB: [wave][buf] V tiles
  __shared__ __bf16 sm[2][16 * 32];          //  2 KiB: per-wave exp(P) tile

  const int blk  = blockIdx.x;
  const int b    = blk / (kS / 16);
  const int qt   = blk % (kS / 16);
  const int wave = threadIdx.x >> 5;
  const int lane = threadIdx.x & 31;
  const int half = lane >> 4;
  const int col  = lane & 15;

  const __bf16* qbase = qb + ((size_t)b * kS + (size_t)qt * 16) * kD;
  const __bf16* kbb   = kb + (size_t)b * kS * kD;
  const __bf16* vbb   = vT + (size_t)b * kD * kS + (size_t)wave * 128 * kS;
  __bf16* smw = &sm[wave][0];

  const v8f vzero = {0.f, 0.f, 0.f, 0.f, 0.f, 0.f, 0.f, 0.f};

  v16bf qf[kD / 32];                                  // resident Q tile
#pragma unroll
  for (int ks = 0; ks < kD / 32; ++ks)
    qf[ks] = load_frag_a_bf(qbase + ks * 32, kD, lane);

  float m[8], lsum[8];
  v8f   ctx[8];
#pragma unroll
  for (int v = 0; v < 8; ++v) { m[v] = -INFINITY; lsum[v] = 0.f; }
#pragma unroll
  for (int n = 0; n < 8; ++n) ctx[n] = vzero;

  // ---- prologue: DMA tile 0 into buffer 0 ----
  if (wave == 0)
    tdm_load_2d(kbb, lds_off_of(&kbuf[0][0]), kD, 32, kD);
  tdm_load_2d(vbb, lds_off_of(&vbuf[wave][0][0]), 32, 128, kS);
  __builtin_amdgcn_s_wait_tensorcnt((short)0);
  __syncthreads();

  for (int kt = 0; kt < kS; kt += 32) {
    const int buf = (kt >> 5) & 1;

    // ---- issue next tile's DMAs into the other buffer (overlap) ----
    if (kt + 32 < kS) {
      if (wave == 0)
        tdm_load_2d(kbb + (size_t)(kt + 32) * kD,
                    lds_off_of(&kbuf[buf ^ 1][0]), kD, 32, kD);
      tdm_load_2d(vbb + (kt + 32),
                  lds_off_of(&vbuf[wave][buf ^ 1][0]), 32, 128, kS);
    }

    // ---- scores: 16x32 tile = q(16xD) * k(ktile)^T  (K from LDS) ----
    const __bf16* kl = &kbuf[buf][0];
    v8f s0 = vzero, s1 = vzero;
#pragma unroll
    for (int ks = 0; ks < kD / 32; ++ks) {
      v16bf k0 = load_frag_b_bf(kl + ks * 32, kD, lane);
      v16bf k1 = load_frag_b_bf(kl + 16 * kD + ks * 32, kD, lane);
      s0 = wmma_bf16(qf[ks], k0, s0);
      s1 = wmma_bf16(qf[ks], k1, s1);
    }

    // ---- online softmax update (fp32) ----
    float alpha[8];
#pragma unroll
    for (int v = 0; v < 8; ++v) {
      float mt = redmax16(fmaxf(s0[v], s1[v]));
      float mn = fmaxf(m[v], mt);
      alpha[v] = __expf(m[v] - mn);
      m[v] = mn;
      float p0 = __expf(s0[v] - mn);
      float p1 = __expf(s1[v] - mn);
      const int row = half * 8 + v;
      smw[row * 32 + col]      = (__bf16)p0;          // C-layout -> LDS
      smw[row * 32 + 16 + col] = (__bf16)p1;
      lsum[v] = lsum[v] * alpha[v] + redsum16(p0 + p1);
    }
#pragma unroll
    for (int n = 0; n < 8; ++n)
#pragma unroll
      for (int v = 0; v < 8; ++v) ctx[n][v] *= alpha[v];

    __syncthreads();                                   // P tile visible (A layout)
    v16bf pa = load_frag_a_bf(smw, 32, lane);
    const __bf16* vl = &vbuf[wave][buf][0];
#pragma unroll
    for (int n = 0; n < 8; ++n) {
      v16bf vf = load_frag_b_bf(vl + n * 16 * 32, 32, lane);
      ctx[n] = wmma_bf16(pa, vf, ctx[n]);
    }

    // ---- retire next tile's DMAs; publish across waves ----
    __builtin_amdgcn_s_wait_tensorcnt((short)0);
    __syncthreads();
  }

  // ---- normalize and store ctx (bf16 [B,S,D]) ----
#pragma unroll
  for (int n = 0; n < 8; ++n) {
    const int feat = wave * 128 + n * 16 + col;
#pragma unroll
    for (int v = 0; v < 8; ++v) {
      const int s = qt * 16 + half * 8 + v;
      ctxb[((size_t)b * kS + s) * kD + feat] = (__bf16)(ctx[n][v] / lsum[v]);
    }
  }
}

// ---------------------------------------------------------------------------
// Kernel 3: output projection + bias + residual + LayerNorm (fp32 out).
// ---------------------------------------------------------------------------
__global__ __launch_bounds__(32) void out_kernel(
    const __bf16* __restrict__ ctxb, const float* __restrict__ x,
    const __bf16* __restrict__ Wd, const float* __restrict__ bd,
    const float* __restrict__ g, const float* __restrict__ bln,
    float* __restrict__ out) {
  const int blk  = blockIdx.x;
  const int b    = blk / (kS / 16);
  const int st   = blk % (kS / 16);
  const int lane = threadIdx.x & 31;
  const int half = lane >> 4;
  const int col  = lane & 15;

  const __bf16* cbase = ctxb + ((size_t)b * kS + (size_t)st * 16) * kD;
  const v8f vzero = {0.f, 0.f, 0.f, 0.f, 0.f, 0.f, 0.f, 0.f};

  v8f acc[kD / 16];
#pragma unroll
  for (int e = 0; e < kD / 16; ++e) acc[e] = vzero;

#pragma unroll
  for (int ks = 0; ks < kD / 32; ++ks) {
    v16bf cf = load_frag_a_bf(cbase + ks * 32, kD, lane);
#pragma unroll
    for (int e = 0; e < kD / 16; ++e) {
      v16bf wf = load_frag_b_bf(Wd + (size_t)(e * 16) * kD + ks * 32, kD, lane);
      acc[e] = wmma_bf16(cf, wf, acc[e]);
    }
  }

  // bias + residual, accumulate per-row moments
  float psum[8], psq[8];
#pragma unroll
  for (int v = 0; v < 8; ++v) { psum[v] = 0.f; psq[v] = 0.f; }
#pragma unroll
  for (int e = 0; e < kD / 16; ++e) {
    const float bde = bd[e * 16 + col];
#pragma unroll
    for (int v = 0; v < 8; ++v) {
      const int s = st * 16 + half * 8 + v;
      float h = acc[e][v] + bde + x[((size_t)b * kS + s) * kD + e * 16 + col];
      acc[e][v] = h;
      psum[v] += h;
      psq[v]  += h * h;
    }
  }

  float mu[8], rstd[8];
#pragma unroll
  for (int v = 0; v < 8; ++v) {
    float sm = redsum16(psum[v]);
    float sq = redsum16(psq[v]);
    mu[v]   = sm * (1.f / kD);
    float var = sq * (1.f / kD) - mu[v] * mu[v];
    rstd[v] = rsqrtf(var + kEps);
  }

#pragma unroll
  for (int e = 0; e < kD / 16; ++e) {
    const int   feat = e * 16 + col;
    const float ge = g[feat];
    const float be = bln[feat];
#pragma unroll
    for (int v = 0; v < 8; ++v) {
      const int s = st * 16 + half * 8 + v;
      out[((size_t)b * kS + s) * kD + feat] = (acc[e][v] - mu[v]) * rstd[v] * ge + be;
    }
  }
}

// ---------------------------------------------------------------------------
// Launch: conversion prep + 3-stage pipeline, all on `stream`.
// ws layout: x_bf | q_bf | k_bf | vT_bf | ctx_bf | Wq_bf | Wk_bf | Wv_bf | Wd_bf
// ---------------------------------------------------------------------------
extern "C" void kernel_launch(void* const* d_in, const int* in_sizes, int n_in,
                              void* d_out, int out_size, void* d_ws, size_t ws_size,
                              hipStream_t stream) {
  (void)in_sizes; (void)n_in; (void)out_size; (void)ws_size;
  const float* x  = (const float*)d_in[0];
  const float* Wq = (const float*)d_in[1];
  const float* bq = (const float*)d_in[2];
  const float* Wk = (const float*)d_in[3];
  const float* bk = (const float*)d_in[4];
  const float* Wv = (const float*)d_in[5];
  const float* bv = (const float*)d_in[6];
  const float* Wd = (const float*)d_in[7];
  const float* bd = (const float*)d_in[8];
  const float* g  = (const float*)d_in[9];
  const float* bl = (const float*)d_in[10];
  float* out = (float*)d_out;

  const size_t nBSD = (size_t)kB * kS * kD;   // 4,194,304
  const size_t nDD  = (size_t)kD * kD;        // 65,536
  __bf16* xbf  = (__bf16*)d_ws;
  __bf16* qbf  = xbf + nBSD;
  __bf16* kbf  = qbf + nBSD;
  __bf16* vT   = kbf + nBSD;
  __bf16* ctxb = vT + nBSD;
  __bf16* Wqb  = ctxb + nBSD;
  __bf16* Wkb  = Wqb + nDD;
  __bf16* Wvb  = Wkb + nDD;
  __bf16* Wdb  = Wvb + nDD;

  // one-time bf16 conversions (x: 8 MiB, weights: 128 KiB each)
  {
    const int n8x = (int)(nBSD / 8);
    cvt_bf16_kernel<<<(n8x + 255) / 256, 256, 0, stream>>>(x, xbf, n8x);
    const int n8w = (int)(nDD / 8);
    const int gw  = (n8w + 255) / 256;
    cvt_bf16_kernel<<<gw, 256, 0, stream>>>(Wq, Wqb, n8w);
    cvt_bf16_kernel<<<gw, 256, 0, stream>>>(Wk, Wkb, n8w);
    cvt_bf16_kernel<<<gw, 256, 0, stream>>>(Wv, Wvb, n8w);
    cvt_bf16_kernel<<<gw, 256, 0, stream>>>(Wd, Wdb, n8w);
  }

  const int tiles = kB * (kS / 16);   // 1024 s-tiles
  qkv_kernel <<<tiles, 128, 0, stream>>>(xbf, Wqb, bq, Wkb, bk, Wvb, bv, qbf, kbf, vT);
  attn_kernel<<<tiles,  64, 0, stream>>>(qbf, kbf, vT, ctxb);
  out_kernel <<<tiles,  32, 0, stream>>>(ctxb, x, Wdb, bd, g, bl, out);
}
